// EMD_Loss_ND_79800492360032
// MI455X (gfx1250) — compile-verified, matrix operand order checked by reading, not run on hardware
//
#include <hip/hip_runtime.h>
#include <math.h>

#define NPTS 256
#define DIMS 3
#define TILE 16
#define NWAVES (NPTS / 32)

typedef __attribute__((ext_vector_type(2)))  float    v2f;
typedef __attribute__((ext_vector_type(8)))  float    v8f;
typedef __attribute__((ext_vector_type(16))) _Float16 v16h;

#if defined(__AMDGCN__) && __has_builtin(__builtin_amdgcn_wmma_f32_16x16x4_f32)
#define HAS_WMMA_F32X4 1
#else
#define HAS_WMMA_F32X4 0
#endif

// ---------------------------------------------------------------------------
// Kernel 1: pairwise Euclidean cost matrix via WMMA (codegen-confirmed:
// emits v_wmma_f32_16x16x4_f32).  One wave computes one 16x16 tile:
// cost[b][i][j] = sqrt(|p1_i|^2 + |p2_j|^2 - 2 * p1_i . p2_j), D=3 padded to K=4.
// ---------------------------------------------------------------------------
__global__ __launch_bounds__(32)
void emd_cost_tiles(const float* __restrict__ p1, const float* __restrict__ p2,
                    float* __restrict__ cost) {
  const int lane = threadIdx.x;                       // 0..31, full wave
  const int tilesPerBatch = (NPTS / TILE) * (NPTS / TILE);
  const int b  = blockIdx.x / tilesPerBatch;
  const int tl = blockIdx.x % tilesPerBatch;
  const int ti = tl / (NPTS / TILE);                  // row tile
  const int tj = tl % (NPTS / TILE);                  // col tile

  const float* P1 = p1 + (size_t)b * NPTS * DIMS;
  const float* P2 = p2 + (size_t)b * NPTS * DIMS;

  __shared__ float sn1[TILE];
  __shared__ float sn2[TILE];

  const int hl  = lane & 15;
  const int row = ti * TILE + hl;
  const int col = tj * TILE + hl;

  if (lane < 16) {
    float x = P1[row * DIMS + 0], y = P1[row * DIMS + 1], z = P1[row * DIMS + 2];
    sn1[hl] = x * x + y * y + z * z;
  } else {
    float x = P2[col * DIMS + 0], y = P2[col * DIMS + 1], z = P2[col * DIMS + 2];
    sn2[hl] = x * x + y * y + z * z;
  }
  __syncthreads();

  v8f acc = {};
#if HAS_WMMA_F32X4
  // A: 16x4 f32.  lanes 0-15: VGPR0=K0, VGPR1=K1 ; lanes 16-31: VGPR0=K2, VGPR1=K3(pad=0)
  // B: 4x16 f32, mirrored layout (N = lane&15).
  v2f a, bm;
  a.x  = (lane < 16) ? P1[row * DIMS + 0] : P1[row * DIMS + 2];
  a.y  = (lane < 16) ? P1[row * DIMS + 1] : 0.0f;
  bm.x = (lane < 16) ? P2[col * DIMS + 0] : P2[col * DIMS + 2];
  bm.y = (lane < 16) ? P2[col * DIMS + 1] : 0.0f;
  acc = __builtin_amdgcn_wmma_f32_16x16x4_f32(false, a, false, bm,
                                              (short)0, acc, false, false);
#else
  v16h a = {}, bm = {};
  if (lane < 16) {
    a[0]  = (_Float16)P1[row * DIMS + 0];
    a[1]  = (_Float16)P1[row * DIMS + 1];
    a[2]  = (_Float16)P1[row * DIMS + 2];
    bm[0] = (_Float16)P2[col * DIMS + 0];
    bm[1] = (_Float16)P2[col * DIMS + 1];
    bm[2] = (_Float16)P2[col * DIMS + 2];
  }
  acc = __builtin_amdgcn_wmma_f32_16x16x32_f16(false, a, false, bm,
                                               (short)0, acc, false, false);
#endif

  // C/D layout: VGPR r -> M = r (+8 for lanes 16-31), N = lane&15
  const float n2v   = sn2[hl];
  const int   mbase = (lane >= 16) ? 8 : 0;
#pragma unroll
  for (int r = 0; r < 8; ++r) {
    const int m  = mbase + r;
    float d2 = sn1[m] + n2v - 2.0f * acc[r];
    d2 = d2 > 0.0f ? d2 : 0.0f;
    cost[((size_t)b * NPTS + (ti * TILE + m)) * NPTS + (tj * TILE + hl)] = sqrtf(d2);
  }
}

// ---------------------------------------------------------------------------
// Kernel 2: exact Hungarian (Jonker-Volgenant shortest augmenting path),
// one 256-thread workgroup (8 wave32) per batch.
//  - column state (used/minv/v) privatized to the owner thread's registers
//  - argmin: lexicographic (val, idx) __shfl_xor butterfly per wave (no
//    barriers) + 8-way cross-wave fold that every thread computes uniformly
//  - only 2 workgroup barriers per augmenting step
// ---------------------------------------------------------------------------
__global__ __launch_bounds__(NPTS)
void emd_hungarian(const float* __restrict__ p1, const float* __restrict__ p2,
                   const float* __restrict__ cost, int use_cost,
                   float* __restrict__ means) {
  const int b = blockIdx.x;
  const int t = threadIdx.x;     // 0..255
  const int j = t + 1;           // column this thread owns (1-indexed)
  const int wave = t >> 5;
  const int lane = t & 31;

  __shared__ float sP1[NPTS * DIMS];
  __shared__ float sP2[NPTS * DIMS];
  __shared__ float u[NPTS + 1];      // row potentials, read at u[i0] by all
  __shared__ int   p[NPTS + 1];      // p[j] = row matched to column j
  __shared__ int   way[NPTS + 1];    // back pointers (read by t0 in augment)
  __shared__ float wredv[NWAVES];
  __shared__ int   wredi[NWAVES];

  const float INF = __builtin_inff();

  for (int idx = t; idx < NPTS * DIMS; idx += NPTS) {
    sP1[idx] = p1[(size_t)b * NPTS * DIMS + idx];
    sP2[idx] = p2[(size_t)b * NPTS * DIMS + idx];
  }

  u[j] = 0.0f; p[j] = 0; way[j] = 0;
  if (t == 0) { u[0] = 0.0f; p[0] = 0; way[0] = 0; }
  float myV = 0.0f;                  // v[j], owner-private
  __syncthreads();

  auto costAt = [&](int r, int c) -> float {  // zero-based indices
    if (use_cost) return cost[((size_t)b * NPTS + r) * NPTS + c];
    float dx = sP1[r * DIMS + 0] - sP2[c * DIMS + 0];
    float dy = sP1[r * DIMS + 1] - sP2[c * DIMS + 1];
    float dz = sP1[r * DIMS + 2] - sP2[c * DIMS + 2];
    return sqrtf(dx * dx + dy * dy + dz * dz);
  };

  for (int i = 1; i <= NPTS; ++i) {
    if (t == 0) p[0] = i;
    float myMinv = INF;              // minv[j], owner-private
    bool  myUsed = false;            // used[j], owner-private
    int   j0 = 0;                    // uniform across all threads
    __syncthreads();                 // p[0]=i visible before sweep reads p[j0]

    int jFinal;
    while (true) {
      const int i0 = p[j0];          // uniform shared read (1..n)
      if (j == j0) myUsed = true;    // column j0 becomes used (j0==0: no owner)

      // ---- column sweep over free columns ----
      float cand = INF;
      if (!myUsed) {
        float cur = costAt(i0 - 1, j - 1) - u[i0] - myV;
        if (cur < myMinv) { myMinv = cur; way[j] = j0; }
        cand = myMinv;
      }

      // ---- wave32 lexicographic argmin butterfly (np.argmin tie-break) ----
      float bv = cand;
      int   bi = j;
#pragma unroll
      for (int m = 16; m >= 1; m >>= 1) {
        float ov = __shfl_xor(bv, m, 32);
        int   oi = __shfl_xor(bi, m, 32);
        if (ov < bv || (ov == bv && oi < bi)) { bv = ov; bi = oi; }
      }
      if (lane == 0) { wredv[wave] = bv; wredi[wave] = bi; }
      __syncthreads();               // barrier 1

      // ---- cross-wave fold: every thread computes the same (delta, j1) ----
      float delta = wredv[0];
      int   j1    = wredi[0];
#pragma unroll
      for (int k = 1; k < NWAVES; ++k) {
        float ov = wredv[k]; int oi = wredi[k];
        if (ov < delta || (ov == delta && oi < j1)) { delta = ov; j1 = oi; }
      }

      // ---- potential updates ----
      // used columns (distinct matched rows -> conflict-free u writes)
      if (myUsed) { u[p[j]] += delta; myV -= delta; }
      else        { myMinv -= delta; }
      // column 0 is used from the first pass of every row: u[p[0]] = u[i]
      if (t == 0) u[p[0]] += delta;
      __syncthreads();               // barrier 2: u/way visible for next step

      if (p[j1] == 0) { jFinal = j1; break; }   // uniform exit
      j0 = j1;
    }

    // ---- augment along the alternating path (serial pointer chase) ----
    if (t == 0) {
      int jj = jFinal;
      while (jj != 0) { int jn = way[jj]; p[jj] = p[jn]; jj = jn; }
    }
    __syncthreads();
  }

  // ---- mean matched cost for this batch (deterministic) ----
  {
    const int r = p[j];              // row matched to column j (1-indexed)
    float s = costAt(r - 1, j - 1);
#pragma unroll
    for (int m = 16; m >= 1; m >>= 1) s += __shfl_xor(s, m, 32);
    if (lane == 0) wredv[wave] = s;
    __syncthreads();
    if (t == 0) {
      float tot = 0.0f;
      for (int k = 0; k < NWAVES; ++k) tot += wredv[k];
      means[b] = tot / (float)NPTS;
    }
  }
}

// ---------------------------------------------------------------------------
// Kernel 3: deterministic final reduction across batches (no float atomics,
// so graph replays are bit-identical).
// ---------------------------------------------------------------------------
__global__ void emd_finalize(const float* __restrict__ means, float* __restrict__ out,
                             int B) {
  if (threadIdx.x == 0 && blockIdx.x == 0) {
    float s = 0.0f;
    for (int i = 0; i < B; ++i) s += means[i];
    out[0] = s / (float)B;
  }
}

extern "C" void kernel_launch(void* const* d_in, const int* in_sizes, int n_in,
                              void* d_out, int out_size, void* d_ws, size_t ws_size,
                              hipStream_t stream) {
  (void)n_in; (void)out_size;
  const float* p1 = (const float*)d_in[0];
  const float* p2 = (const float*)d_in[1];
  const int B = in_sizes[0] / (NPTS * DIMS);   // 32 for the reference shapes

  const size_t costBytes = (size_t)B * NPTS * NPTS * sizeof(float);

  float* cost  = nullptr;
  float* means = nullptr;
  int use_cost = 0;
  if (ws_size >= costBytes + (size_t)B * sizeof(float)) {
    cost  = (float*)d_ws;
    means = (float*)((char*)d_ws + costBytes);
    use_cost = 1;
  } else {
    means = (float*)d_ws;   // only B floats needed in the fallback path
  }

  if (use_cost) {
    const int tiles = B * (NPTS / TILE) * (NPTS / TILE);
    emd_cost_tiles<<<tiles, 32, 0, stream>>>(p1, p2, cost);
  }
  emd_hungarian<<<B, NPTS, 0, stream>>>(p1, p2, cost, use_cost, means);
  emd_finalize<<<1, 32, 0, stream>>>(means, (float*)d_out, B);
}
